// MultiHeadSelfAttention_80891414053241
// MI455X (gfx1250) — matrix operand order repaired and checked by reading, not compiled, not run on hardware
//
#include <hip/hip_runtime.h>
#include <hip/hip_bf16.h>

typedef __attribute__((ext_vector_type(16))) __bf16 v16bf;
typedef __attribute__((ext_vector_type(8)))  __bf16 v8bf;
typedef __attribute__((ext_vector_type(8)))  float  v8f;
typedef int v4i_ __attribute__((vector_size(16)));

#define B_  4
#define T_  2048
#define C_  1024
#define H_  16
#define DH_ 64

__device__ __forceinline__ unsigned short f2bf(float f) {
  unsigned u = __builtin_bit_cast(unsigned, f);
  u += 0x7FFFu + ((u >> 16) & 1u);          // round-to-nearest-even
  return (unsigned short)(u >> 16);
}
__device__ __forceinline__ unsigned pk2(float a, float b) {
  return (unsigned)f2bf(a) | ((unsigned)f2bf(b) << 16);
}
__device__ __forceinline__ v16bf cat8(v8bf lo, v8bf hi) {
  return __builtin_shufflevector(lo, hi, 0,1,2,3,4,5,6,7,8,9,10,11,12,13,14,15);
}

// ----- CDNA5 async global->LDS copy (16B per lane, ASYNCcnt tracked) -------
__device__ __forceinline__ void async_ld16(const void* g, void* l) {
#if __has_builtin(__builtin_amdgcn_global_load_async_to_lds_b128)
  typedef __attribute__((address_space(1))) v4i_* gp_t;   // global int4*
  typedef __attribute__((address_space(3))) v4i_* lp_t;   // LDS int4*
  __builtin_amdgcn_global_load_async_to_lds_b128(
      (gp_t)(size_t)g,            // inttoptr: generic global addr == AS1 addr
      (lp_t)(v4i_*)l,             // addrspacecast generic -> AS3
      0, 0);
#else
  unsigned lo = (unsigned)(size_t)l;                 // LDS byte offset = addr[31:0]
  asm volatile("global_load_async_to_lds_b128 %0, %1, off"
               :: "v"(lo), "v"(g) : "memory");
#endif
}
__device__ __forceinline__ void wait_async0() {
#if __has_builtin(__builtin_amdgcn_s_wait_asynccnt)
  __builtin_amdgcn_s_wait_asynccnt(0);
#else
  asm volatile("s_wait_asynccnt 0" ::: "memory");
#endif
}

// ---------------------------------------------------------------------------
// Tiled bf16 GEMM:  [8192 x 1024] x [1024 x N]   (N = 3072 for QKV, 1024 proj)
// Block: 256 thr (8 waves), tile 128(M) x 128(N) x 32(K); wave = 2x4 WMMA tiles
// MODE 0: A = fp32 x (convert on stage-in), epilogue scatters bf16 QKV[3][B][H][T][64]
// MODE 1: A = bf16 attn-out (async copy to LDS), epilogue writes fp32 d_out
// ---------------------------------------------------------------------------
template<int MODE>
__global__ __launch_bounds__(256) void gemm_k(const float* __restrict__ Af,
                                              const unsigned short* __restrict__ Ah,
                                              const float* __restrict__ W,
                                              unsigned short* __restrict__ qkv,
                                              float* __restrict__ fout)
{
  constexpr int N = (MODE == 0) ? 3 * C_ : C_;
  constexpr int K = C_;
  __shared__ __align__(16) unsigned short As[128 * 40];   // [m][k], stride 40
  __shared__ __align__(16) unsigned short Bs[128 * 40];   // transposed [n][k]

  const int tid  = threadIdx.x;
  const int wave = tid >> 5, lane = tid & 31;
  const int ln = lane & 15, lh = lane >> 4;
  const int wm = wave >> 1, wn = wave & 1;
  const int m0 = blockIdx.y * 128;
  const int n0 = blockIdx.x * 128;

  v8f zero = {};
  v8f acc[2][4];
  #pragma unroll
  for (int mi = 0; mi < 2; ++mi)
    #pragma unroll
    for (int ni = 0; ni < 4; ++ni) acc[mi][ni] = zero;

  for (int kb = 0; kb < K; kb += 32) {
    // ---- stage A tile 128x32 ----
    if constexpr (MODE == 0) {
      #pragma unroll
      for (int i = 0; i < 4; ++i) {
        int c = tid + i * 256;
        int row = c >> 3, kc = (c & 7) * 4;
        const float4 v = *(const float4*)(Af + (size_t)(m0 + row) * K + kb + kc);
        *(unsigned*)(As + row * 40 + kc)     = pk2(v.x, v.y);
        *(unsigned*)(As + row * 40 + kc + 2) = pk2(v.z, v.w);
      }
    } else {
      #pragma unroll
      for (int i = 0; i < 2; ++i) {            // async bf16 copy, 16B per lane
        int c = tid + i * 256;
        int row = c >> 2, kc8 = (c & 3) * 8;
        async_ld16(Ah + (size_t)(m0 + row) * K + kb + kc8, As + row * 40 + kc8);
      }
    }
    // ---- stage B tile 32x128 fp32 -> transposed bf16 [n][k] ----
    #pragma unroll
    for (int i = 0; i < 4; ++i) {
      int c = tid + i * 256;
      int k = c >> 5, nc = (c & 31) * 4;
      const float4 v = *(const float4*)(W + (size_t)(kb + k) * N + n0 + nc);
      Bs[(nc + 0) * 40 + k] = f2bf(v.x);
      Bs[(nc + 1) * 40 + k] = f2bf(v.y);
      Bs[(nc + 2) * 40 + k] = f2bf(v.z);
      Bs[(nc + 3) * 40 + k] = f2bf(v.w);
    }
    if (kb + 32 < K) {                          // prefetch next K-slab
      __builtin_prefetch(W + (size_t)(kb + 32 + (tid >> 5)) * N + n0 + (tid & 31) * 4, 0, 1);
      if constexpr (MODE == 0)
        __builtin_prefetch(Af + (size_t)(m0 + (tid >> 1)) * K + kb + 32 + (tid & 1) * 16, 0, 1);
    }
    if constexpr (MODE == 1) wait_async0();
    __syncthreads();

    v16bf af[2], bf[4];
    #pragma unroll
    for (int mi = 0; mi < 2; ++mi) {            // A frag: row m, K in table order
      const unsigned short* p = As + (wm * 32 + mi * 16 + ln) * 40;
      af[mi] = cat8(*(const v8bf*)(p + lh * 8), *(const v8bf*)(p + 16 + lh * 8));
    }
    #pragma unroll
    for (int ni = 0; ni < 4; ++ni) {            // B frag: col n, K in table order
      const unsigned short* p = Bs + (wn * 64 + ni * 16 + ln) * 40;
      bf[ni] = cat8(*(const v8bf*)(p + lh * 8), *(const v8bf*)(p + 16 + lh * 8));
    }
    #pragma unroll
    for (int mi = 0; mi < 2; ++mi)
      #pragma unroll
      for (int ni = 0; ni < 4; ++ni)
        acc[mi][ni] = __builtin_amdgcn_wmma_f32_16x16x32_bf16(
            false, af[mi], false, bf[ni], (short)0, acc[mi][ni], false, false);
    __syncthreads();
  }

  // epilogue (C layout: lane col n = ln, VGPR r -> row r + 8*lh)
  #pragma unroll
  for (int mi = 0; mi < 2; ++mi)
    #pragma unroll
    for (int ni = 0; ni < 4; ++ni)
      #pragma unroll
      for (int r = 0; r < 8; ++r) {
        int gm = m0 + wm * 32 + mi * 16 + r + 8 * lh;
        int gn = n0 + wn * 64 + ni * 16 + ln;
        float v = acc[mi][ni][r];
        if constexpr (MODE == 0) {
          int s  = gn >> 10;              // 0=Q 1=K 2=V
          int hd = (gn >> 6) & (H_ - 1);  // head
          int dh = gn & (DH_ - 1);
          int b  = gm >> 11;              // / T_
          int t  = gm & (T_ - 1);
          size_t off = ((size_t)(s * B_ * H_ + b * H_ + hd) * T_ + t) * DH_ + dh;
          qkv[off] = f2bf(v);
        } else {
          fout[(size_t)gm * C_ + gn] = v;
        }
      }
}

// ---------------------------------------------------------------------------
// Flash attention: block = (b, head, 128 queries); 8 waves x 16 query rows.
// Online softmax in exp2 domain; K tile async-staged to LDS, V transposed.
// ---------------------------------------------------------------------------
__global__ __launch_bounds__(256) void attn_k(const unsigned short* __restrict__ qkv,
                                              unsigned short* __restrict__ aout)
{
  __shared__ __align__(16) unsigned short Ks[32 * 72];     // [key][dh]
  __shared__ __align__(16) unsigned short Vt[64 * 40];     // transposed [dh][key]
  __shared__ __align__(16) unsigned short Pw[8][16 * 40];  // per-wave P scratch

  const int b  = blockIdx.z;
  const int hd = blockIdx.y;
  const int q0 = blockIdx.x * 128;
  const int tid  = threadIdx.x;
  const int wave = tid >> 5, lane = tid & 31;
  const int ln = lane & 15, lh = lane >> 4;
  const int qw = q0 + wave * 16;

  const size_t plane = (size_t)(b * H_ + hd) * T_ * DH_;
  const size_t SZ = (size_t)B_ * H_ * T_ * DH_;
  const unsigned short* Qb = qkv + plane;
  const unsigned short* Kb = qkv + SZ + plane;
  const unsigned short* Vb = qkv + 2 * SZ + plane;

  // Q A-fragments (row qw+ln), two 32-wide dh chunks, loaded once
  v16bf qf[2];
  {
    const unsigned short* qr = Qb + (size_t)(qw + ln) * DH_;
    #pragma unroll
    for (int c = 0; c < 2; ++c)
      qf[c] = cat8(*(const v8bf*)(qr + c * 32 + lh * 8),
                   *(const v8bf*)(qr + c * 32 + 16 + lh * 8));
  }

  v8f zero = {};
  v8f o[4]; o[0] = zero; o[1] = zero; o[2] = zero; o[3] = zero;
  float mrun[8], lsum[8];
  #pragma unroll
  for (int r = 0; r < 8; ++r) { mrun[r] = -1e30f; lsum[r] = 0.f; }

  const float SC = 0.125f * 1.44269504088896f;   // 1/sqrt(Dh) * log2(e)
  unsigned short* pw = &Pw[wave][0];
  const int kend = q0 + 128;                     // causal: keys <= last query of block

  for (int k0 = 0; k0 < kend; k0 += 32) {
    { // stage 32-key K/V tiles: K via async-to-LDS, V transposed manually
      int row = tid >> 3, c8 = (tid & 7) * 8;
      async_ld16(Kb + (size_t)(k0 + row) * DH_ + c8, Ks + row * 72 + c8);
      uint4 vv = *(const uint4*)(Vb + (size_t)(k0 + row) * DH_ + c8);
      Vt[(c8 + 0) * 40 + row] = (unsigned short)(vv.x);
      Vt[(c8 + 1) * 40 + row] = (unsigned short)(vv.x >> 16);
      Vt[(c8 + 2) * 40 + row] = (unsigned short)(vv.y);
      Vt[(c8 + 3) * 40 + row] = (unsigned short)(vv.y >> 16);
      Vt[(c8 + 4) * 40 + row] = (unsigned short)(vv.z);
      Vt[(c8 + 5) * 40 + row] = (unsigned short)(vv.z >> 16);
      Vt[(c8 + 6) * 40 + row] = (unsigned short)(vv.w);
      Vt[(c8 + 7) * 40 + row] = (unsigned short)(vv.w >> 16);
      if (k0 + 32 < kend) {  // prefetch next tile -> global_prefetch_b8
        __builtin_prefetch(Kb + (size_t)(k0 + 32 + row) * DH_ + c8, 0, 1);
        __builtin_prefetch(Vb + (size_t)(k0 + 32 + row) * DH_ + c8, 0, 1);
      }
    }
    wait_async0();
    __syncthreads();

    if (k0 < qw + 16) {   // wave-uniform causal skip
      // S = Q K^T : two 16-key score tiles, K-frag rows are contiguous K-rows
      v8f s0 = zero, s1 = zero;
      #pragma unroll
      for (int c = 0; c < 2; ++c) {
        const unsigned short* p0 = Ks + ln * 72 + c * 32;
        v16bf kf0 = cat8(*(const v8bf*)(p0 + lh * 8), *(const v8bf*)(p0 + 16 + lh * 8));
        s0 = __builtin_amdgcn_wmma_f32_16x16x32_bf16(false, qf[c], false, kf0, (short)0, s0, false, false);
        const unsigned short* p1 = Ks + (16 + ln) * 72 + c * 32;
        v16bf kf1 = cat8(*(const v8bf*)(p1 + lh * 8), *(const v8bf*)(p1 + 16 + lh * 8));
        s1 = __builtin_amdgcn_wmma_f32_16x16x32_bf16(false, qf[c], false, kf1, (short)0, s1, false, false);
      }
      // online softmax (per-row stats live in element r for row r+8*lh)
      #pragma unroll
      for (int r = 0; r < 8; ++r) {
        int q  = qw + r + 8 * lh;
        int c0 = k0 + ln, c1 = c0 + 16;
        float x0 = (c0 <= q) ? s0[r] * SC : -1e30f;
        float x1 = (c1 <= q) ? s1[r] * SC : -1e30f;
        float rm = fmaxf(x0, x1);
        #pragma unroll
        for (int msk = 1; msk < 16; msk <<= 1)
          rm = fmaxf(rm, __shfl_xor(rm, msk, 32));
        float mn = fmaxf(mrun[r], rm);
        float al = exp2f(mrun[r] - mn);
        mrun[r] = mn;
        float e0 = exp2f(x0 - mn);
        float e1 = exp2f(x1 - mn);
        float rs = e0 + e1;
        #pragma unroll
        for (int msk = 1; msk < 16; msk <<= 1)
          rs += __shfl_xor(rs, msk, 32);
        lsum[r] = lsum[r] * al + rs;
        o[0][r] *= al; o[1][r] *= al; o[2][r] *= al; o[3][r] *= al;
        pw[(r + 8 * lh) * 40 + ln]      = f2bf(e0);  // C-layout -> A-layout via LDS
        pw[(r + 8 * lh) * 40 + 16 + ln] = f2bf(e1);
      }
      // O += P V  (P A-frag from wave-private LDS; V B-frag from transposed LDS)
      const unsigned short* pp = pw + ln * 40;
      v16bf pf = cat8(*(const v8bf*)(pp + lh * 8), *(const v8bf*)(pp + 16 + lh * 8));
      #pragma unroll
      for (int t = 0; t < 4; ++t) {
        const unsigned short* vp = Vt + (t * 16 + ln) * 40;
        v16bf vf = cat8(*(const v8bf*)(vp + lh * 8), *(const v8bf*)(vp + 16 + lh * 8));
        o[t] = __builtin_amdgcn_wmma_f32_16x16x32_bf16(false, pf, false, vf, (short)0, o[t], false, false);
      }
    }
    __syncthreads();
  }

  // normalize + write bf16 [B*T, C] (heads concatenated)
  #pragma unroll
  for (int r = 0; r < 8; ++r) {
    float inv = 1.0f / lsum[r];
    int q = qw + r + 8 * lh;
    size_t rowoff = (size_t)(b * T_ + q) * C_ + hd * DH_;
    #pragma unroll
    for (int t = 0; t < 4; ++t)
      aout[rowoff + t * 16 + ln] = f2bf(o[t][r] * inv);
  }
}

// ---------------------------------------------------------------------------
extern "C" void kernel_launch(void* const* d_in, const int* in_sizes, int n_in,
                              void* d_out, int out_size, void* d_ws, size_t ws_size,
                              hipStream_t stream) {
  (void)in_sizes; (void)n_in; (void)out_size; (void)ws_size;
  const float* x  = (const float*)d_in[0];   // [B,T,1024]
  const float* wq = (const float*)d_in[1];   // [1024,3072]
  const float* wp = (const float*)d_in[2];   // [1024,1024]
  float* out = (float*)d_out;                // [B,T,1024]

  unsigned short* qkv  = (unsigned short*)d_ws;                 // 48 MB bf16 [3,B,H,T,64]
  unsigned short* aout = qkv + (size_t)3 * B_ * H_ * T_ * DH_;  // 16 MB bf16 [B*T, C]

  dim3 blk(256);
  gemm_k<0><<<dim3((3 * C_) / 128, (B_ * T_) / 128), blk, 0, stream>>>(x, nullptr, wq, qkv, nullptr);
  attn_k<<<dim3(T_ / 128, H_, B_), blk, 0, stream>>>(qkv, aout);
  gemm_k<1><<<dim3(C_ / 128, (B_ * T_) / 128), blk, 0, stream>>>(nullptr, aout, wp, nullptr, out);
}
